// Head_63015760166916
// MI455X (gfx1250) — matrix operand order repaired and checked by reading, not compiled, not run on hardware
//
#include <hip/hip_runtime.h>

// ---------------------------------------------------------------------------
// MI455X (gfx1250) fused attention:
//   proj = x @ w  (fp32 in, bf16 WMMA w/ fp32 acc)  -> q*scale, k, vT (bf16)
//   out  = softmax(q k^T) v   (flash-attention, v_wmma_f32_16x16x32_bf16)
// Workspace (bf16): wT 384x1024 | q 8x4096x128 | k 8x4096x128 | vT 8x128x4096
// Attention: 8 waves/block share K/V tiles through double-buffered LDS
// (one cooperative global load per block instead of 8 redundant L2 reads);
// sched_barrier(0) pins load batches ahead of WMMA bursts so waits are partial.
// ---------------------------------------------------------------------------

typedef __attribute__((ext_vector_type(16))) __bf16 v16bf;
typedef __attribute__((ext_vector_type(8)))  float  v8f;

union AB16 {
    v16bf          v;
    unsigned short u[16];
    unsigned int   d[8];
    uint4          q2[2];
};

__device__ __forceinline__ unsigned short f2bf(float f) {
    unsigned int u = __float_as_uint(f);
    u += 0x7FFFu + ((u >> 16) & 1u);       // round-to-nearest-even
    return (unsigned short)(u >> 16);
}

// Packed 2xf32 -> 2xbf16 (RNE), single VALU op (assembler-verified on gfx1250).
__device__ __forceinline__ unsigned int cvt_pk_bf16(float lo, float hi) {
    unsigned int d;
    asm("v_cvt_pk_bf16_f32 %0, %1, %2" : "=v"(d) : "v"(lo), "v"(hi));
    return d;
}

__device__ __forceinline__ v8f wmma_bf16(v16bf a, v16bf b, v8f c) {
    return __builtin_amdgcn_wmma_f32_16x16x32_bf16(
        false, a, false, b, (short)0, c, false, false);
}

#define SEQ 4096
#define DMODEL 1024
#define DH 128
#define NBATCH 8

// ---------------------------------------------------------------------------
// Kernel 0: w (1024 x 384 fp32, row-major) -> wT (384 x 1024 bf16, k-contig)
// ---------------------------------------------------------------------------
__global__ __launch_bounds__(256) void convert_w(const float* __restrict__ w,
                                                 unsigned short* __restrict__ wT) {
    int idx = blockIdx.x * 256 + threadIdx.x;           // over 384*1024
    int n  = idx >> 10;
    int kk = idx & 1023;
    wT[idx] = f2bf(w[(size_t)kk * 384 + n]);
}

// ---------------------------------------------------------------------------
// Kernel 1: projection GEMM, 8-way A reuse, pinned-pipelined B loads.
// Wave = 16-row M tile x 128-col N group (ng: 0=q, 1=k, 2=v).
// ---------------------------------------------------------------------------
__global__ __launch_bounds__(256) void proj_kernel(
    const float* __restrict__ x, const unsigned short* __restrict__ wT,
    unsigned short* __restrict__ qg, unsigned short* __restrict__ kg,
    unsigned short* __restrict__ vg) {
    const int wid  = blockIdx.x * 8 + (threadIdx.x >> 5);   // 0..6143
    const int lane = threadIdx.x & 31;
    const int hf   = lane >> 4;
    const int col  = lane & 15;
    const int mt   = wid / 3;                               // 0..2047
    const int ng   = wid % 3;                               // 0=q, 1=k, 2=v

    const float*          xr = x  + (size_t)(mt * 16 + col) * DMODEL;   // A row
    const unsigned short* wb = wT + (size_t)(ng * DH + col) * DMODEL;   // B col base

    v8f acc[8];
#pragma unroll
    for (int g = 0; g < 8; ++g) acc[g] = {};

    for (int ks = 0; ks < DMODEL; ks += 32) {
        // ---- batch-issue all loads for this K-step ----
        AB16 Bp[8];
#pragma unroll
        for (int g = 0; g < 8; ++g) {
            const unsigned short* pb = wb + (size_t)g * 16 * DMODEL + ks + hf * 8;
            Bp[g].q2[0] = *(const uint4*)(pb);
            Bp[g].q2[1] = *(const uint4*)(pb + 16);
        }
        const float* pa = xr + ks + hf * 8;
        float4 f0 = *(const float4*)(pa);
        float4 f1 = *(const float4*)(pa + 4);
        float4 f2 = *(const float4*)(pa + 16);
        float4 f3 = *(const float4*)(pa + 20);
        __builtin_amdgcn_sched_barrier(0);   // keep loads hoisted above WMMAs

        AB16 A;
        A.d[0] = cvt_pk_bf16(f0.x, f0.y);
        A.d[1] = cvt_pk_bf16(f0.z, f0.w);
        A.d[2] = cvt_pk_bf16(f1.x, f1.y);
        A.d[3] = cvt_pk_bf16(f1.z, f1.w);
        A.d[4] = cvt_pk_bf16(f2.x, f2.y);
        A.d[5] = cvt_pk_bf16(f2.z, f2.w);
        A.d[6] = cvt_pk_bf16(f3.x, f3.y);
        A.d[7] = cvt_pk_bf16(f3.z, f3.w);

        // ---- WMMA burst (waits become partial s_wait_loadcnt) ----
#pragma unroll
        for (int g = 0; g < 8; ++g) acc[g] = wmma_bf16(A.v, Bp[g].v, acc[g]);
    }

    const float scale = 0.08838834764831845f;   // 1/sqrt(128), folded into q
    if (ng == 0) {
#pragma unroll
        for (int g = 0; g < 8; ++g)
#pragma unroll
            for (int r = 0; r < 8; ++r) {
                const int grow = mt * 16 + hf * 8 + r;
                qg[(size_t)grow * DH + g * 16 + col] = f2bf(acc[g][r] * scale);
            }
    } else if (ng == 1) {
#pragma unroll
        for (int g = 0; g < 8; ++g)
#pragma unroll
            for (int r = 0; r < 8; ++r) {
                const int grow = mt * 16 + hf * 8 + r;
                kg[(size_t)grow * DH + g * 16 + col] = f2bf(acc[g][r]);
            }
    } else {
#pragma unroll
        for (int g = 0; g < 8; ++g)
#pragma unroll
            for (int r = 0; r < 8; ++r) {
                const int grow = mt * 16 + hf * 8 + r;
                const int bb = grow >> 12, ss = grow & 4095;
                vg[((size_t)bb * DH + g * 16 + col) * SEQ + ss] = f2bf(acc[g][r]);
            }
    }
}

// ---------------------------------------------------------------------------
// Kernel 2: flash attention. Block = 256 thr (8 waves) = 128 q-rows.
// Per 32-key step: K/V tiles cooperatively staged in double-buffered LDS;
// 8 score WMMAs + fp32 online softmax + 8 p@v WMMAs per wave.
// ---------------------------------------------------------------------------
__global__ __launch_bounds__(256) void attn_kernel(
    const unsigned short* __restrict__ qg, const unsigned short* __restrict__ kg,
    const unsigned short* __restrict__ vg, float* __restrict__ out) {
    __shared__ __align__(32) unsigned short Kb[2][32 * DH];    // [key][dh]
    __shared__ __align__(32) unsigned short Vb[2][DH * 32];    // [dh][key]
    __shared__ __align__(32) unsigned short Plds[8][16 * 32];  // per-wave P

    const int b    = blockIdx.x;
    const int qt   = blockIdx.y;
    const int tid  = threadIdx.x;
    const int wave = tid >> 5;
    const int lane = tid & 31;
    const int hf   = lane >> 4;
    const int col  = lane & 15;

    const int qrow0 = qt * 128 + wave * 16;
    const unsigned short* qb = qg + ((size_t)b * SEQ + qrow0) * DH;
    const unsigned short* kb = kg + (size_t)b * SEQ * DH;
    const unsigned short* vb = vg + (size_t)b * DH * SEQ;

    // cooperative-staging indices: K tile 32x128 (32B per thread), V tile 128x32
    const int krow = tid >> 3, kcg = tid & 7;      // key row, 16-dh col group
    const int vrow = tid >> 1, vcg = tid & 1;      // dh row, 16-key col group

    // Q tile (16 x 128) resident in registers as 4 A-operands (16x32 each)
    AB16 Aq[4];
#pragma unroll
    for (int c = 0; c < 4; ++c) {
        const unsigned short* p = qb + (size_t)col * DH + c * 32 + hf * 8;
        Aq[c].q2[0] = *(const uint4*)(p);
        Aq[c].q2[1] = *(const uint4*)(p + 16);
    }

    v8f acc[8];
#pragma unroll
    for (int g = 0; g < 8; ++g) acc[g] = {};
    float mi[8], li[8];
#pragma unroll
    for (int r = 0; r < 8; ++r) { mi[r] = -__builtin_inff(); li[r] = 0.f; }

    unsigned short* pl = &Plds[wave][0];

    // ---- preload tile 0 into LDS buffer 0 ----
    {
        const unsigned short* ks = kb + (size_t)krow * DH + kcg * 16;
        uint4 a0 = *(const uint4*)(ks);
        uint4 a1 = *(const uint4*)(ks + 8);
        const unsigned short* vs = vb + (size_t)vrow * SEQ + vcg * 16;
        uint4 b0 = *(const uint4*)(vs);
        uint4 b1 = *(const uint4*)(vs + 8);
        *(uint4*)(&Kb[0][krow * DH + kcg * 16])     = a0;
        *(uint4*)(&Kb[0][krow * DH + kcg * 16 + 8]) = a1;
        *(uint4*)(&Vb[0][vrow * 32 + vcg * 16])     = b0;
        *(uint4*)(&Vb[0][vrow * 32 + vcg * 16 + 8]) = b1;
    }
    __syncthreads();

    for (int it = 0; it < SEQ / 32; ++it) {
        const int cur = it & 1;
        const int ktn = ((it + 1) * 32) & (SEQ - 1);   // wraps in-bounds on last iter

        // ---- issue next tile's global loads; compute below hides latency ----
        uint4 ka0, ka1, va0, va1;
        {
            const unsigned short* ks = kb + (size_t)(ktn + krow) * DH + kcg * 16;
            ka0 = *(const uint4*)(ks);
            ka1 = *(const uint4*)(ks + 8);
            const unsigned short* vs = vb + (size_t)vrow * SEQ + ktn + vcg * 16;
            va0 = *(const uint4*)(vs);
            va1 = *(const uint4*)(vs + 8);
        }

        const unsigned short* Kc = &Kb[cur][0];
        const unsigned short* Vc = &Vb[cur][0];

        // ---- batch K B-operand ds loads, then 8 score WMMAs ----
        AB16 Bk0[4], Bk1[4];
#pragma unroll
        for (int c = 0; c < 4; ++c) {
            const unsigned short* p0 = Kc + col * DH + c * 32 + hf * 8;
            Bk0[c].q2[0] = *(const uint4*)(p0);
            Bk0[c].q2[1] = *(const uint4*)(p0 + 16);
            const unsigned short* p1 = p0 + 16 * DH;
            Bk1[c].q2[0] = *(const uint4*)(p1);
            Bk1[c].q2[1] = *(const uint4*)(p1 + 16);
        }
        __builtin_amdgcn_sched_barrier(0);

        v8f s0 = {}, s1 = {};
#pragma unroll
        for (int c = 0; c < 4; ++c) {
            s0 = wmma_bf16(Aq[c].v, Bk0[c].v, s0);
            s1 = wmma_bf16(Aq[c].v, Bk1[c].v, s1);
        }

        // ---- batch V B-operand ds loads; softmax hides their latency ----
        AB16 Bv[8];
#pragma unroll
        for (int g = 0; g < 8; ++g) {
            const unsigned short* pv = Vc + (g * 16 + col) * 32 + hf * 8;
            Bv[g].q2[0] = *(const uint4*)(pv);
            Bv[g].q2[1] = *(const uint4*)(pv + 16);
        }
        __builtin_amdgcn_sched_barrier(0);

        // ---- fp32 online softmax (rows = hf*8+r, half-wave shfl reduce) ----
        float al[8], p0f[8], p1f[8];
#pragma unroll
        for (int r = 0; r < 8; ++r) {
            float t = fmaxf(s0[r], s1[r]);
            t = fmaxf(t, __shfl_xor(t, 1, 32));
            t = fmaxf(t, __shfl_xor(t, 2, 32));
            t = fmaxf(t, __shfl_xor(t, 4, 32));
            t = fmaxf(t, __shfl_xor(t, 8, 32));
            const float mnew = fmaxf(mi[r], t);
            p0f[r] = __expf(s0[r] - mnew);
            p1f[r] = __expf(s1[r] - mnew);
            float rs = p0f[r] + p1f[r];
            rs += __shfl_xor(rs, 1, 32);
            rs += __shfl_xor(rs, 2, 32);
            rs += __shfl_xor(rs, 4, 32);
            rs += __shfl_xor(rs, 8, 32);
            al[r] = __expf(mi[r] - mnew);
            li[r] = li[r] * al[r] + rs;
            mi[r] = mnew;
        }
#pragma unroll
        for (int g = 0; g < 8; ++g)
#pragma unroll
            for (int r = 0; r < 8; ++r) acc[g][r] *= al[r];

        // ---- P: C layout -> LDS (row-major 16x32 bf16) -> A layout ----
#pragma unroll
        for (int r = 0; r < 8; ++r) {
            const int row = hf * 8 + r;
            pl[row * 32 + col]      = f2bf(p0f[r]);
            pl[row * 32 + 16 + col] = f2bf(p1f[r]);
        }
        asm volatile("s_wait_dscnt 0" ::: "memory");   // ds_store -> ds_load
        AB16 Pa;
        {
            const unsigned short* pp = pl + col * 32 + hf * 8;
            Pa.q2[0] = *(const uint4*)(pp);
            Pa.q2[1] = *(const uint4*)(pp + 16);
        }
        asm volatile("" ::: "memory");

        // ---- out += P @ V : 8 dh-groups (V already in registers) ----
#pragma unroll
        for (int g = 0; g < 8; ++g) acc[g] = wmma_bf16(Pa.v, Bv[g].v, acc[g]);

        // ---- rotate double buffer ----
        __syncthreads();                               // all waves done reading
        const int nxt = cur ^ 1;
        *(uint4*)(&Kb[nxt][krow * DH + kcg * 16])     = ka0;
        *(uint4*)(&Kb[nxt][krow * DH + kcg * 16 + 8]) = ka1;
        *(uint4*)(&Vb[nxt][vrow * 32 + vcg * 16])     = va0;
        *(uint4*)(&Vb[nxt][vrow * 32 + vcg * 16 + 8]) = va1;
        __syncthreads();                               // staged data visible
    }

    // ---- epilogue: out = acc / l ----
#pragma unroll
    for (int g = 0; g < 8; ++g)
#pragma unroll
        for (int r = 0; r < 8; ++r) {
            const size_t o = ((size_t)b * SEQ + qrow0 + hf * 8 + r) * DH + g * 16 + col;
            out[o] = acc[g][r] / li[r];
        }
}

// ---------------------------------------------------------------------------
extern "C" void kernel_launch(void* const* d_in, const int* in_sizes, int n_in,
                              void* d_out, int out_size, void* d_ws, size_t ws_size,
                              hipStream_t stream) {
    const float* x = (const float*)d_in[0];
    const float* w = (const float*)d_in[1];
    float* out = (float*)d_out;

    unsigned short* wT = (unsigned short*)d_ws;                    // 384*1024
    unsigned short* qg = wT + (size_t)384 * 1024;                  // 8*4096*128
    unsigned short* kg = qg + (size_t)NBATCH * SEQ * DH;
    unsigned short* vg = kg + (size_t)NBATCH * SEQ * DH;           // vT

    convert_w<<<(384 * 1024) / 256, 256, 0, stream>>>(w, wT);
    // 2048 M-tiles * 3 N-groups = 6144 waves / 8 waves per block
    proj_kernel<<<768, 256, 0, stream>>>(x, wT, qg, kg, vg);
    dim3 grid(NBATCH, SEQ / 128);                                  // 256 blocks x 8 waves
    attn_kernel<<<grid, 256, 0, stream>>>(qg, kg, vg, out);
}
